// MultiheadAttention_46531675684993
// MI455X (gfx1250) — compile-verified, matrix operand order checked by reading, not run on hardware
//
#include <hip/hip_runtime.h>
#include <hip/hip_bf16.h>

// ---------------------------------------------------------------------------
// MI455X (gfx1250) fused "buggy MHA" pipeline, bf16 WMMA everywhere.
// Stages:
//   0. fp32 -> bf16 conversion of x, Wk, Wv, Wo
//   1. K = Xbf . Wk^T   (WMMA GEMM, bf16 out)
//   2. V = Xbf . Wv^T   (WMMA GEMM, bf16 out)
//   3. Vt = transpose of V per (b,h): [b,h,hd,s]
//   4. fused  scores = V.K^T (masked)  ->  ctx = scores.V  ->  softmax(head_dim)
//   5. out = Ctx . Wo^T + bo  (WMMA GEMM, fp32 out)
// ---------------------------------------------------------------------------

typedef __attribute__((ext_vector_type(16))) __bf16 bf16x16;
typedef __attribute__((ext_vector_type(8)))  __bf16 bf16x8;
typedef __attribute__((ext_vector_type(8)))  float  f32x8;

__device__ __forceinline__ __bf16 f2bf(float f) {
  union { float f; unsigned u; } a; a.f = f;
  unsigned r = a.u + 0x7FFFu + ((a.u >> 16) & 1u);   // round-to-nearest-even
  union { unsigned short s; __bf16 b; } o; o.s = (unsigned short)(r >> 16);
  return o.b;
}

__device__ __forceinline__ f32x8 zero8() {
  f32x8 z = {0.f,0.f,0.f,0.f,0.f,0.f,0.f,0.f};
  return z;
}

// Assemble a 16x32 bf16 WMMA operand fragment (per-lane 16 values) from two
// 16-byte chunks: K = {kb..kb+7} and K = {kb+16..kb+23} for this lane.
__device__ __forceinline__ bf16x16 load_frag(const __bf16* p0, const __bf16* p1) {
  bf16x8 lo = *(const bf16x8*)p0;
  bf16x8 hi = *(const bf16x8*)p1;
  return __builtin_shufflevector(lo, hi, 0,1,2,3,4,5,6,7,8,9,10,11,12,13,14,15);
}

// ---------------------------------------------------------------------------
// fp32 -> bf16 conversion (vectorized by 4)
// ---------------------------------------------------------------------------
__global__ __launch_bounds__(256)
void cvt_bf16_kernel(const float* __restrict__ in, __bf16* __restrict__ out, int n4) {
  int i = blockIdx.x * 256 + threadIdx.x;
  if (i < n4) {
    float4 v = ((const float4*)in)[i];
    out[i*4+0] = f2bf(v.x);
    out[i*4+1] = f2bf(v.y);
    out[i*4+2] = f2bf(v.z);
    out[i*4+3] = f2bf(v.w);
  }
}

// ---------------------------------------------------------------------------
// Tiled WMMA GEMM:  C[M,N] = A[M,K] . B[N,K]^T   (A,B bf16 row-major)
// Block tile 128x64, K-step 64 through LDS, 8 waves of 32 (wave32),
// each wave owns a 32x32 micro-tile = 2x2 v_wmma_f32_16x16x32_bf16 accums.
// Output either bf16 (intermediate) or fp32 (+bias) for the final stage.
// ---------------------------------------------------------------------------
__global__ __launch_bounds__(256)
void gemm_bf16_kernel(const __bf16* __restrict__ A, const __bf16* __restrict__ Bm,
                      int M, int N, int K,
                      __bf16* __restrict__ outBF, float* __restrict__ outF,
                      const float* __restrict__ bias) {
  __shared__ __align__(16) __bf16 As[128*64];
  __shared__ __align__(16) __bf16 Bs[64*64];

  const int tid  = threadIdx.x;
  const int lane = tid & 31;
  const int wave = tid >> 5;
  const int wm   = wave & 3;            // wave M position (0..3)
  const int wn   = wave >> 2;           // wave N position (0..1)
  const int bm   = blockIdx.y * 128;
  const int bn   = blockIdx.x * 64;
  const int ln   = lane & 15;
  const int kb   = (lane >> 4) << 3;    // 0 or 8 (fragment K sub-offset)

  f32x8 acc[2][2];
  #pragma unroll
  for (int i = 0; i < 2; ++i)
    #pragma unroll
    for (int j = 0; j < 2; ++j)
      acc[i][j] = zero8();

  for (int k0 = 0; k0 < K; k0 += 64) {
    // Stage A tile (128x64) and B tile (64x64) into LDS as 16B chunks.
    #pragma unroll
    for (int c = 0; c < 4; ++c) {
      int chunk = tid + c * 256;
      int r = chunk >> 3, cc = (chunk & 7) << 3;
      *(bf16x8*)&As[r*64 + cc] = *(const bf16x8*)&A[(size_t)(bm + r)*K + k0 + cc];
    }
    #pragma unroll
    for (int c = 0; c < 2; ++c) {
      int chunk = tid + c * 256;
      int r = chunk >> 3, cc = (chunk & 7) << 3;
      *(bf16x8*)&Bs[r*64 + cc] = *(const bf16x8*)&Bm[(size_t)(bn + r)*K + k0 + cc];
    }
    if (k0 + 64 < K) {   // CDNA5 global_prefetch_b8 of the next K tile
      __builtin_prefetch(&A[(size_t)(bm + (tid & 127))*K + k0 + 64], 0, 3);
      __builtin_prefetch(&Bm[(size_t)(bn + (tid & 63))*K + k0 + 64], 0, 3);
    }
    __syncthreads();

    #pragma unroll
    for (int kk = 0; kk < 64; kk += 32) {
      bf16x16 af[2], bg[2];
      #pragma unroll
      for (int i = 0; i < 2; ++i) {
        int r = wm*32 + i*16 + ln;
        af[i] = load_frag(&As[r*64 + kk + kb], &As[r*64 + kk + kb + 16]);
      }
      #pragma unroll
      for (int j = 0; j < 2; ++j) {
        int r = wn*32 + j*16 + ln;
        bg[j] = load_frag(&Bs[r*64 + kk + kb], &Bs[r*64 + kk + kb + 16]);
      }
      #pragma unroll
      for (int i = 0; i < 2; ++i)
        #pragma unroll
        for (int j = 0; j < 2; ++j)
          acc[i][j] = __builtin_amdgcn_wmma_f32_16x16x32_bf16(
              false, af[i], false, bg[j], (short)0, acc[i][j], false, false);
    }
    __syncthreads();
  }

  // Epilogue. C/D layout: VGPR v -> M=v (lanes 0-15) or M=v+8 (lanes 16-31),
  // N = lane&15.
  const int rsel = (lane >> 4) << 3;
  #pragma unroll
  for (int i = 0; i < 2; ++i)
    #pragma unroll
    for (int j = 0; j < 2; ++j)
      #pragma unroll
      for (int v = 0; v < 8; ++v) {
        int row = bm + wm*32 + i*16 + v + rsel;
        int col = bn + wn*32 + j*16 + ln;
        float val = acc[i][j][v];
        if (outBF) outBF[(size_t)row*N + col] = f2bf(val);
        else       outF [(size_t)row*N + col] = val + (bias ? bias[col] : 0.f);
      }
}

// ---------------------------------------------------------------------------
// V[b,s,h,hd] (row-major (B*S,2048)) -> Vt[b,h,hd,s]  (bf16 32x32 LDS tiles)
// ---------------------------------------------------------------------------
__global__ __launch_bounds__(256)
void transpose_v_kernel(const __bf16* __restrict__ V, __bf16* __restrict__ Vt) {
  __shared__ __bf16 tile[32][33];
  const int bh = blockIdx.z;            // b*16 + h
  const int s0 = blockIdx.y * 32;
  const int d0 = blockIdx.x * 32;       // within head_dim
  const int b  = bh >> 4, h = bh & 15;
  const int tx = threadIdx.x & 31;
  const int ty = threadIdx.x >> 5;      // 0..7
  #pragma unroll
  for (int i = 0; i < 4; ++i) {
    int s = s0 + ty + i*8;
    tile[ty + i*8][tx] = V[((size_t)(b*2048 + s))*2048 + h*128 + d0 + tx];
  }
  __syncthreads();
  #pragma unroll
  for (int i = 0; i < 4; ++i) {
    int d = d0 + ty + i*8;
    Vt[((size_t)(bh*128 + d))*2048 + s0 + tx] = tile[tx][ty + i*8];
  }
}

// ---------------------------------------------------------------------------
// Fused attention middle: per (b,h,16-query rows) one wave computes
//   scores(16x32) = Vq(16x128) . K(t:t+32,128)^T      (4 WMMA k-steps x 2)
//   causal mask (t > s -> -inf)
//   ctx(16x128) += scores(16x32) . V(t:t+32, 128)     (8 WMMA n-tiles)
// then softmax over head_dim per row, bf16 out in (B,H,S,HD) flat order.
// ---------------------------------------------------------------------------
__global__ __launch_bounds__(32)
void attn_kernel(const __bf16* __restrict__ Kb, const __bf16* __restrict__ Vb,
                 const __bf16* __restrict__ Vt, __bf16* __restrict__ Ctx) {
  constexpr int S = 2048, E = 2048, HD = 128;
  __shared__ __align__(16) __bf16 sS[16*32];   // bf16 score tile
  __shared__ float sC[16*128];                  // fp32 ctx tile

  const int qb   = blockIdx.x & (S/16 - 1);
  const int bh   = blockIdx.x / (S/16);
  const int b    = bh >> 4, h = bh & 15;
  const int s0   = qb * 16;
  const int lane = threadIdx.x;
  const int ln   = lane & 15;
  const int kb   = (lane >> 4) << 3;

  const __bf16* Vrow = Vb + (size_t)b*S*E + h*HD;   // + s*E + hd
  const __bf16* Krow = Kb + (size_t)b*S*E + h*HD;
  const __bf16* VtB  = Vt + (size_t)bh*HD*S;        // + hd*S + t

  // A-fragments of Vq: 16x128 as 4 k-step fragments (resident in VGPRs).
  bf16x16 vq[4];
  #pragma unroll
  for (int f = 0; f < 4; ++f) {
    const __bf16* p = Vrow + (size_t)(s0 + ln)*E + f*32 + kb;
    vq[f] = load_frag(p, p + 16);
  }

  f32x8 acc[8];
  #pragma unroll
  for (int n = 0; n < 8; ++n) acc[n] = zero8();

  const int nt = (s0 + 16 + 31) >> 5;     // causal: only t-blocks that matter
  for (int it = 0; it < nt; ++it) {
    const int t0 = it * 32;

    f32x8 sc[2];
    #pragma unroll
    for (int sub = 0; sub < 2; ++sub) sc[sub] = zero8();
    #pragma unroll
    for (int sub = 0; sub < 2; ++sub) {
      const __bf16* kp = Krow + (size_t)(t0 + sub*16 + ln)*E;
      #pragma unroll
      for (int f = 0; f < 4; ++f) {
        bf16x16 kf = load_frag(kp + f*32 + kb, kp + f*32 + kb + 16);
        sc[sub] = __builtin_amdgcn_wmma_f32_16x16x32_bf16(
            false, vq[f], false, kf, (short)0, sc[sub], false, false);
      }
    }

    // Causal mask in-register on the WMMA C layout.
    #pragma unroll
    for (int sub = 0; sub < 2; ++sub)
      #pragma unroll
      for (int v = 0; v < 8; ++v) {
        int t = t0 + sub*16 + ln;
        int s = s0 + v + kb;
        if (t > s) sc[sub][v] = -__builtin_inff();
      }

    __syncthreads();
    #pragma unroll
    for (int sub = 0; sub < 2; ++sub)
      #pragma unroll
      for (int v = 0; v < 8; ++v)
        sS[(v + kb)*32 + sub*16 + ln] = f2bf(sc[sub][v]);
    __syncthreads();

    // Re-fragment scores as WMMA A operand, multiply by V block (from Vt).
    bf16x16 sa = load_frag(&sS[ln*32 + kb], &sS[ln*32 + kb + 16]);
    #pragma unroll
    for (int n = 0; n < 8; ++n) {
      const __bf16* vp = VtB + (size_t)(n*16 + ln)*S + t0;
      bf16x16 vf = load_frag(vp + kb, vp + kb + 16);
      acc[n] = __builtin_amdgcn_wmma_f32_16x16x32_bf16(
          false, sa, false, vf, (short)0, acc[n], false, false);
    }
  }

  // Spill ctx tile to LDS, then per-row softmax over head_dim (128).
  __syncthreads();
  #pragma unroll
  for (int n = 0; n < 8; ++n)
    #pragma unroll
    for (int v = 0; v < 8; ++v)
      sC[(v + kb)*128 + n*16 + ln] = acc[n][v];
  __syncthreads();

  if (lane < 16) {
    const float scale = 0.08838834764831845f;   // 1/sqrt(128)
    float mx = -__builtin_inff();
    for (int c = 0; c < 128; ++c) mx = fmaxf(mx, sC[lane*128 + c] * scale);
    float sum = 0.f;
    for (int c = 0; c < 128; ++c) sum += __expf(sC[lane*128 + c]*scale - mx);
    float inv = 1.f / sum;
    __bf16* out = Ctx + ((size_t)bh * S + s0 + lane) * HD;   // flat (B,H,S,HD)
    for (int c = 0; c < 128; ++c)
      out[c] = f2bf(__expf(sC[lane*128 + c]*scale - mx) * inv);
  }
}

// ---------------------------------------------------------------------------
extern "C" void kernel_launch(void* const* d_in, const int* in_sizes, int n_in,
                              void* d_out, int out_size, void* d_ws, size_t ws_size,
                              hipStream_t stream) {
  (void)in_sizes; (void)n_in; (void)out_size; (void)ws_size;
  const float* x  = (const float*)d_in[0];
  // d_in[1] = Wq: dead code in the reference, skipped.
  const float* Wk = (const float*)d_in[2];
  const float* Wv = (const float*)d_in[3];
  const float* Wo = (const float*)d_in[4];
  const float* bo = (const float*)d_in[5];

  constexpr size_t B = 4, S = 2048, E = 2048;
  constexpr size_t MX = B * S;  // 8192

  char* ws = (char*)d_ws;
  size_t off = 0;
  auto alloc = [&](size_t bytes) {
    void* p = ws + off;
    off += (bytes + 255) & ~(size_t)255;
    return p;
  };
  __bf16* Xbf = (__bf16*)alloc(MX*E*2);
  __bf16* Wkb = (__bf16*)alloc(E*E*2);
  __bf16* Wvb = (__bf16*)alloc(E*E*2);
  __bf16* Wob = (__bf16*)alloc(E*E*2);
  __bf16* Kbf = (__bf16*)alloc(MX*E*2);
  __bf16* Vbf = (__bf16*)alloc(MX*E*2);
  __bf16* Vtb = (__bf16*)alloc(MX*E*2);
  __bf16* Ctx = (__bf16*)alloc(MX*E*2);

  auto cvt = [&](const float* in, __bf16* out, size_t n) {
    int n4 = (int)(n / 4);
    cvt_bf16_kernel<<<(n4 + 255) / 256, 256, 0, stream>>>(in, out, n4);
  };
  cvt(x,  Xbf, MX*E);
  cvt(Wk, Wkb, E*E);
  cvt(Wv, Wvb, E*E);
  cvt(Wo, Wob, E*E);

  dim3 gg(E/64, MX/128);   // (32, 64)
  gemm_bf16_kernel<<<gg, 256, 0, stream>>>(Xbf, Wkb, (int)MX, (int)E, (int)E,
                                           Kbf, nullptr, nullptr);
  gemm_bf16_kernel<<<gg, 256, 0, stream>>>(Xbf, Wvb, (int)MX, (int)E, (int)E,
                                           Vbf, nullptr, nullptr);

  dim3 tg(4, S/32, B*16);  // (HD/32, S/32, B*H)
  transpose_v_kernel<<<tg, 256, 0, stream>>>(Vbf, Vtb);

  attn_kernel<<<(int)(B*16*(S/16)), 32, 0, stream>>>(Kbf, Vbf, Vtb, Ctx);

  // Ctx flat (B,H,S,HD) reinterpreted as (B*S, E) row-major == reference's
  // buggy reshape. Final GEMM with bias to fp32 output.
  gemm_bf16_kernel<<<gg, 256, 0, stream>>>(Ctx, Wob, (int)MX, (int)E, (int)E,
                                           nullptr, (float*)d_out, bo);
}
